// Pix_Projector2_59760174957011
// MI455X (gfx1250) — matrix operand order repaired and checked by reading, not compiled, not run on hardware
//
#include <hip/hip_runtime.h>
#include <hip/hip_bf16.h>

typedef __attribute__((ext_vector_type(16))) _Float16 v16h;
typedef __attribute__((ext_vector_type(8)))  float    v8f;

#define CIN   256
#define CMID  512
#define COUT  128
#define HW    6400          // 80*80
#define NB    16
#define TILE_N 32
#define OUTS  160

// ---- workspace layout (bytes) ----
#define OFF_W1  0
#define OFF_W2  (512*256*2)
#define OFF_W3  (OFF_W2 + 256*512*2)
#define OFF_F2  (OFF_W3 + 128*256*2)                       // 589824, 32B aligned
#define OFF_ACC (OFF_F2 + (size_t)NB*HW*COUT*2)            // f2: 26,214,400 B
#define OFF_CNT (OFF_ACC + (size_t)NB*8*COUT*4)

// -------------------------------------------------------------------------
// Pack f32 row-major W[M][K] into WMMA 16x16x32 A-operand order:
// per 16x32 tile, 32 lanes x 16 halfs contiguous.
//   lane<16 : K = j                            (K in {0..15})
//   lane>=16: K = j + 16                       (K in {16..31})
// -------------------------------------------------------------------------
__global__ void pack_w(const float* __restrict__ W, _Float16* __restrict__ dst,
                       int M, int K) {
  int e = blockIdx.x * 256 + threadIdx.x;
  if (e >= M * K) return;
  int t    = e >> 9;            // tile index (512 halfs per tile)
  int r    = e & 511;
  int lane = r >> 4;
  int j    = r & 15;
  int kTiles = K >> 5;
  int mT = t / kTiles, kT = t % kTiles;
  int m  = mT * 16 + (lane & 15);
  int koff = j + ((lane >> 4) << 4);
  int k  = kT * 32 + koff;
  dst[e] = (_Float16)W[(size_t)m * K + k];
}

__global__ void zero_k(float* __restrict__ p, int n) {
  int i = blockIdx.x * 256 + threadIdx.x;
  if (i < n) p[i] = 0.0f;
}

// -------------------------------------------------------------------------
// Fused 3-layer 1x1-conv MLP over a tile of 32 pixels, all-WMMA.
// LDS: X[32][256]f16 (16KB) | Y1[32][512]f16 (32KB) | Y2[32][256]f16 (16KB)
// 256 threads = 8 wave32; activations stay in LDS, only f2 goes to HBM (f16).
// -------------------------------------------------------------------------
__global__ void __launch_bounds__(256)
mlp_fused(const float* __restrict__ feat,
          const _Float16* __restrict__ W1, const _Float16* __restrict__ W2,
          const _Float16* __restrict__ W3,
          const float* __restrict__ b1, const float* __restrict__ b2,
          const float* __restrict__ b3,
          _Float16* __restrict__ f2out) {
  extern __shared__ _Float16 smem[];
  _Float16* Xs = smem;                 // 32*256
  _Float16* Y1 = smem + TILE_N * CIN;  // 32*512
  _Float16* Y2 = Y1 + TILE_N * CMID;   // 32*256

  const int tid  = threadIdx.x;
  const int lane = tid & 31;
  const int wave = tid >> 5;
  const int p0   = blockIdx.x * TILE_N;          // global pixel base
  const int b    = p0 / HW;
  const int hw0  = p0 % HW;

  // Stage X tile: coalesced along hw within each channel, f32 -> f16.
  for (int i = tid; i < TILE_N * CIN; i += 256) {
    int c = i >> 5, pix = i & 31;
    Xs[pix * CIN + c] = (_Float16)feat[((size_t)b * CIN + c) * HW + hw0 + pix];
  }
  __syncthreads();

  const int lhk  = (lane >> 4) << 4;   // B-operand K offset: 0 or 16
  const int pixl = lane & 15;          // D/B column within tile
  const int mhi  = (lane >> 4) << 3;   // D row offset: 0 or 8

  // ---------------- Layer 1: Y1 = relu(W1[512x256] * X + b1) --------------
  for (int mi = 0; mi < 4; ++mi) {
    int mT = wave * 4 + mi;
    v8f a0 = {}, a1 = {};
    for (int kt = 0; kt < CIN / 32; ++kt) {
      v16h A  = ((const v16h*)W1)[(mT * (CIN / 32) + kt) * 32 + lane];
      v16h B0 = *(const v16h*)(Xs + (pixl) * CIN + kt * 32 + lhk);
      v16h B1 = *(const v16h*)(Xs + (16 + pixl) * CIN + kt * 32 + lhk);
      a0 = __builtin_amdgcn_wmma_f32_16x16x32_f16(false, A, false, B0, (short)0, a0, false, false);
      a1 = __builtin_amdgcn_wmma_f32_16x16x32_f16(false, A, false, B1, (short)0, a1, false, false);
    }
    int mrow = mT * 16 + mhi;
    for (int r = 0; r < 8; ++r) {
      int m = mrow + r;
      float bb = b1[m];
      float v0 = a0[r] + bb; v0 = v0 > 0.f ? v0 : 0.f;
      float v1 = a1[r] + bb; v1 = v1 > 0.f ? v1 : 0.f;
      Y1[(pixl) * CMID + m]      = (_Float16)v0;
      Y1[(16 + pixl) * CMID + m] = (_Float16)v1;
    }
  }
  __syncthreads();

  // ---------------- Layer 2: Y2 = relu(W2[256x512] * Y1 + b2) -------------
  for (int mi = 0; mi < 2; ++mi) {
    int mT = wave * 2 + mi;
    v8f a0 = {}, a1 = {};
    for (int kt = 0; kt < CMID / 32; ++kt) {
      v16h A  = ((const v16h*)W2)[(mT * (CMID / 32) + kt) * 32 + lane];
      v16h B0 = *(const v16h*)(Y1 + (pixl) * CMID + kt * 32 + lhk);
      v16h B1 = *(const v16h*)(Y1 + (16 + pixl) * CMID + kt * 32 + lhk);
      a0 = __builtin_amdgcn_wmma_f32_16x16x32_f16(false, A, false, B0, (short)0, a0, false, false);
      a1 = __builtin_amdgcn_wmma_f32_16x16x32_f16(false, A, false, B1, (short)0, a1, false, false);
    }
    int mrow = mT * 16 + mhi;
    for (int r = 0; r < 8; ++r) {
      int m = mrow + r;
      float bb = b2[m];
      float v0 = a0[r] + bb; v0 = v0 > 0.f ? v0 : 0.f;
      float v1 = a1[r] + bb; v1 = v1 > 0.f ? v1 : 0.f;
      Y2[(pixl) * CIN + m]      = (_Float16)v0;
      Y2[(16 + pixl) * CIN + m] = (_Float16)v1;
    }
  }
  __syncthreads();

  // ---------------- Layer 3: f2 = relu(W3[128x256] * Y2 + b3) -> HBM ------
  {
    int mT = wave;  // 8 waves cover the 8 output M-tiles
    v8f a0 = {}, a1 = {};
    for (int kt = 0; kt < CIN / 32; ++kt) {
      v16h A  = ((const v16h*)W3)[(mT * (CIN / 32) + kt) * 32 + lane];
      v16h B0 = *(const v16h*)(Y2 + (pixl) * CIN + kt * 32 + lhk);
      v16h B1 = *(const v16h*)(Y2 + (16 + pixl) * CIN + kt * 32 + lhk);
      a0 = __builtin_amdgcn_wmma_f32_16x16x32_f16(false, A, false, B0, (short)0, a0, false, false);
      a1 = __builtin_amdgcn_wmma_f32_16x16x32_f16(false, A, false, B1, (short)0, a1, false, false);
    }
    int mrow = mT * 16 + mhi;
    size_t gbase = (size_t)p0 * COUT;   // f2 layout: [b][h][w][c], c contiguous
    for (int r = 0; r < 8; ++r) {
      int m = mrow + r;
      float bb = b3[m];
      float v0 = a0[r] + bb; v0 = v0 > 0.f ? v0 : 0.f;
      float v1 = a1[r] + bb; v1 = v1 > 0.f ? v1 : 0.f;
      f2out[gbase + (size_t)(pixl) * COUT + m]      = (_Float16)v0;
      f2out[gbase + (size_t)(16 + pixl) * COUT + m] = (_Float16)v1;
    }
  }
}

// -------------------------------------------------------------------------
// Bilinear 80->160 resize + label-bucket sum. One block per (b, out-row).
// Per-class partial sums in LDS (ds_add_f32), one global atomic flush/block.
// f2 (26 MB, f16) is L2-resident on MI455X (192 MB L2).
// -------------------------------------------------------------------------
__global__ void __launch_bounds__(256)
pool_kernel(const _Float16* __restrict__ f2, const int* __restrict__ lbl,
            float* __restrict__ acc, float* __restrict__ cnt) {
  __shared__ float ssum[8 * COUT];
  __shared__ float scnt[8];
  const int tid = threadIdx.x;
  for (int i = tid; i < 8 * COUT; i += 256) ssum[i] = 0.0f;
  if (tid < 8) scnt[tid] = 0.0f;
  __syncthreads();

  const int b = blockIdx.x / OUTS;
  const int Y = blockIdx.x % OUTS;

  // half-pixel mapping: src = (i+0.5)*(80/160) - 0.5
  float sy = (Y + 0.5f) * 0.5f - 0.5f;
  int   y0 = (int)floorf(sy);
  float wy = sy - (float)y0;
  int y0c = y0 < 0 ? 0 : y0;
  int y1c = (y0 + 1) > 79 ? 79 : (y0 + 1);

  const int c    = tid & 127;
  const int half = tid >> 7;
  const int* lrow = lbl + (size_t)b * OUTS * OUTS + (size_t)Y * OUTS;
  const _Float16* base = f2 + (size_t)b * HW * COUT;

  for (int X = half; X < OUTS; X += 2) {
    float sx = (X + 0.5f) * 0.5f - 0.5f;
    int   x0 = (int)floorf(sx);
    float wx = sx - (float)x0;
    int x0c = x0 < 0 ? 0 : x0;
    int x1c = (x0 + 1) > 79 ? 79 : (x0 + 1);

    float f00 = (float)base[((size_t)y0c * 80 + x0c) * COUT + c];
    float f01 = (float)base[((size_t)y0c * 80 + x1c) * COUT + c];
    float f10 = (float)base[((size_t)y1c * 80 + x0c) * COUT + c];
    float f11 = (float)base[((size_t)y1c * 80 + x1c) * COUT + c];
    float v = (1.f - wy) * ((1.f - wx) * f00 + wx * f01)
            +        wy  * ((1.f - wx) * f10 + wx * f11);

    int k = lrow[X];
    atomicAdd(&ssum[k * COUT + c], v);           // ds_add_f32
    if (c == 0) atomicAdd(&scnt[k], 1.0f);
  }
  __syncthreads();

  float* gacc = acc + (size_t)b * 8 * COUT;
  for (int i = tid; i < 8 * COUT; i += 256) atomicAdd(&gacc[i], ssum[i]);
  if (tid < 8) atomicAdd(&cnt[b * 8 + tid], scnt[tid]);
}

__global__ void finalize_k(const float* __restrict__ acc,
                           const float* __restrict__ cnt,
                           float* __restrict__ out) {
  int tid = blockIdx.x * 256 + threadIdx.x;
  if (tid < NB * 8 * COUT) {
    float c = cnt[tid >> 7];                      // [b][k] matches acc [b][k][c]
    out[tid] = acc[tid] / (c + 1e-8f);
  }
  if (tid < NB * 8) {
    out[NB * 8 * COUT + tid] = cnt[tid] > 0.0f ? 1.0f : 0.0f;
  }
}

// -------------------------------------------------------------------------
extern "C" void kernel_launch(void* const* d_in, const int* in_sizes, int n_in,
                              void* d_out, int out_size, void* d_ws, size_t ws_size,
                              hipStream_t stream) {
  const float* feat = (const float*)d_in[0];
  const int*   lbl  = (const int*)d_in[1];
  const float* w1   = (const float*)d_in[2];
  const float* b1   = (const float*)d_in[3];
  const float* w2   = (const float*)d_in[4];
  const float* b2   = (const float*)d_in[5];
  const float* w3   = (const float*)d_in[6];
  const float* b3   = (const float*)d_in[7];

  char* ws = (char*)d_ws;
  _Float16* pW1 = (_Float16*)(ws + OFF_W1);
  _Float16* pW2 = (_Float16*)(ws + OFF_W2);
  _Float16* pW3 = (_Float16*)(ws + OFF_W3);
  _Float16* f2  = (_Float16*)(ws + OFF_F2);
  float*    acc = (float*)(ws + OFF_ACC);
  float*    cnt = (float*)(ws + OFF_CNT);

  // Pack weights to WMMA A-operand order (f32 -> f16).
  pack_w<<<512, 256, 0, stream>>>(w1, pW1, 512, 256);
  pack_w<<<512, 256, 0, stream>>>(w2, pW2, 256, 512);
  pack_w<<<128, 256, 0, stream>>>(w3, pW3, 128, 256);

  // Zero accumulators every call (graph replay safety).
  zero_k<<<65, 256, 0, stream>>>(acc, NB * 8 * COUT + NB * 8);

  // Fused MLP: 102400 pixels / 32 per block = 3200 blocks, 64KB dynamic LDS.
  int shmem = (TILE_N * CIN + TILE_N * CMID + TILE_N * CIN) * (int)sizeof(_Float16);
  mlp_fused<<<(NB * HW) / TILE_N, 256, shmem, stream>>>(feat, pW1, pW2, pW3,
                                                        b1, b2, b3, f2);

  // Resize + label pooling: one block per (b, output row).
  pool_kernel<<<NB * OUTS, 256, 0, stream>>>(f2, lbl, acc, cnt);

  finalize_k<<<64, 256, 0, stream>>>(acc, cnt, (float*)d_out);
}
